// dgNN_GAT_70918499991573
// MI455X (gfx1250) — matrix operand order repaired and checked by reading, not compiled, not run on hardware
//
#include <hip/hip_runtime.h>
#include <hip/hip_bf16.h>
#include <math.h>

typedef float v8f  __attribute__((ext_vector_type(8)));
typedef _Float16 v16h __attribute__((ext_vector_type(16)));
typedef _Float16 v8h  __attribute__((ext_vector_type(8)));

#define NEG_SLOPE 0.2f

// ---------------------------------------------------------------------------
// float atomic max via integer atomics (works for mixed signs; init = -inf)
// ---------------------------------------------------------------------------
__device__ __forceinline__ void atomicMaxF(float* addr, float val) {
    if (val >= 0.0f) {
        atomicMax((int*)addr, __float_as_int(val));
    } else {
        atomicMin((unsigned int*)addr, __float_as_uint(val));
    }
}

// ---------------------------------------------------------------------------
// GEMM: C[nrows x NCOLS] = A[nrows x 128] * W[128 x NCOLS], fp32 in/out,
// f16 WMMA accumulate-in-f32.  One wave per 16-row strip.
// W staged in LDS *transposed* (Wt[col][k], padded stride) so each B fragment
// is two contiguous 16-byte LDS reads (2 x ds_load_b128) per WMMA.
// ---------------------------------------------------------------------------
template <int NCOLS>
__global__ void __launch_bounds__(256) gemm_wmma(const float* __restrict__ A,
                                                 const float* __restrict__ W,
                                                 float* __restrict__ C,
                                                 int nrows) {
    constexpr int K  = 128;
    constexpr int NT = NCOLS / 16;
    constexpr int KP = K + 8;               // padded column stride (136*2=272B, 16B-aligned)
    __shared__ _Float16 Wt[NCOLS * KP];

    // stage + convert + transpose W (fp32 row-major -> f16 col-major) into LDS
    for (int i = threadIdx.x; i < K * NCOLS; i += blockDim.x) {
        int k = i / NCOLS, c = i % NCOLS;   // global read coalesced over c
        Wt[c * KP + k] = (_Float16)W[i];
    }
    __syncthreads();

    const int wave  = threadIdx.x >> 5;
    const int lane  = threadIdx.x & 31;
    const int strip = blockIdx.x * (blockDim.x >> 5) + wave;
    const int rowBase = strip * 16;
    if (rowBase >= nrows) return;            // whole-wave guard: EXEC stays all-1s

    const int half = lane >> 4;              // 0: lanes 0-15, 1: lanes 16-31
    const int lrow = lane & 15;
    int arow_i = rowBase + lrow;
    if (arow_i >= nrows) arow_i = nrows - 1; // safety clamp (nrows%16==0 normally)
    const float* __restrict__ arow = A + (size_t)arow_i * K;

    v8f acc[NT];
#pragma unroll
    for (int t = 0; t < NT; ++t) acc[t] = (v8f)0.0f;

#pragma unroll
    for (int kb = 0; kb < K; kb += 32) {
        // ---- A fragment: 16-bit A 16x32, contiguous k order (k = 16*half + s) ----
        // lanes 0-15 hold K = kb+0..15; lanes 16-31 hold kb+16..31
        const int koff = kb + half * 16;     // multiple of 16 -> 64B aligned
        const float4 x0 = *(const float4*)(arow + koff);
        const float4 x1 = *(const float4*)(arow + koff + 4);
        const float4 x2 = *(const float4*)(arow + koff + 8);
        const float4 x3 = *(const float4*)(arow + koff + 12);
        v16h af;
        af[0]  = (_Float16)x0.x; af[1]  = (_Float16)x0.y;
        af[2]  = (_Float16)x0.z; af[3]  = (_Float16)x0.w;
        af[4]  = (_Float16)x1.x; af[5]  = (_Float16)x1.y;
        af[6]  = (_Float16)x1.z; af[7]  = (_Float16)x1.w;
        af[8]  = (_Float16)x2.x; af[9]  = (_Float16)x2.y;
        af[10] = (_Float16)x2.z; af[11] = (_Float16)x2.w;
        af[12] = (_Float16)x3.x; af[13] = (_Float16)x3.y;
        af[14] = (_Float16)x3.z; af[15] = (_Float16)x3.w;

        // ---- B fragments: lanes 0-15 hold K = kb+0..15, lanes 16-31 hold kb+16..31
        const int ks = kb + half * 16;       // multiple of 16 -> 32B offset in column
#pragma unroll
        for (int t = 0; t < NT; ++t) {
            const _Float16* __restrict__ wcol = &Wt[(t * 16 + lrow) * KP + ks];
            const v8h lo = *(const v8h*)(wcol);       // K = ks..ks+7   (ds_load_b128)
            const v8h hi = *(const v8h*)(wcol + 8);   // K = ks+8..ks+15 (ds_load_b128)
            const v16h bf = __builtin_shufflevector(lo, hi,
                0, 1, 2, 3, 4, 5, 6, 7, 8, 9, 10, 11, 12, 13, 14, 15);
            acc[t] = __builtin_amdgcn_wmma_f32_16x16x32_f16(
                false, af, false, bf, (short)0, acc[t], false, false);
        }
    }

    // ---- C store: VGPR r -> M = r (lanes 0-15) / r+8 (lanes 16-31), N = lane&15
#pragma unroll
    for (int t = 0; t < NT; ++t) {
#pragma unroll
        for (int r = 0; r < 8; ++r) {
            int row = rowBase + r + half * 8;
            if (row < nrows)
                C[(size_t)row * NCOLS + t * 16 + lrow] = acc[t][r];
        }
    }
}

// ---------------------------------------------------------------------------
// a_src[n,h] = sum_c feat[n,h,c]*att_src[h,c];  a_dst likewise
// ---------------------------------------------------------------------------
template <int H, int C>
__global__ void attn_proj(const float* __restrict__ feat,
                          const float* __restrict__ att_src,
                          const float* __restrict__ att_dst,
                          float* __restrict__ asrc, float* __restrict__ adst,
                          int N) {
    int tid = blockIdx.x * blockDim.x + threadIdx.x;
    if (tid >= N * H) return;
    const int h = tid % H;
    const float* __restrict__ f = feat + (size_t)tid * C;
    float s1 = 0.f, s2 = 0.f;
#pragma unroll
    for (int c = 0; c < C; ++c) {
        float v = f[c];
        s1 += v * att_src[h * C + c];
        s2 += v * att_dst[h * C + c];
    }
    asrc[tid] = s1;
    adst[tid] = s2;
}

// ---------------------------------------------------------------------------
__global__ void fillf(float* p, long long n, float v) {
    long long i = blockIdx.x * (long long)blockDim.x + threadIdx.x;
    if (i < n) p[i] = v;
}

// ---------------------------------------------------------------------------
template <int H>
__global__ void edge_max(const long long* __restrict__ ei, long long E,
                         const float* __restrict__ asrc,
                         const float* __restrict__ adst,
                         float* __restrict__ m) {
    long long i = blockIdx.x * (long long)blockDim.x + threadIdx.x;
    if (i >= E) return;
    const long long s = ei[i], d = ei[E + i];
#pragma unroll
    for (int h = 0; h < H; ++h) {
        float e = asrc[s * H + h] + adst[d * H + h];
        e = e > 0.f ? e : NEG_SLOPE * e;
        atomicMaxF(&m[d * H + h], e);
    }
}

template <int H>
__global__ void edge_sum(const long long* __restrict__ ei, long long E,
                         const float* __restrict__ asrc,
                         const float* __restrict__ adst,
                         const float* __restrict__ m,
                         float* __restrict__ ssum) {
    long long i = blockIdx.x * (long long)blockDim.x + threadIdx.x;
    if (i >= E) return;
    const long long s = ei[i], d = ei[E + i];
#pragma unroll
    for (int h = 0; h < H; ++h) {
        float e = asrc[s * H + h] + adst[d * H + h];
        e = e > 0.f ? e : NEG_SLOPE * e;
        atomicAdd(&ssum[d * H + h], __expf(e - m[d * H + h]));
    }
}

// thread per (edge, head, 4-channel group)
template <int H, int C, int CG>
__global__ void edge_aggregate(const long long* __restrict__ ei, long long E,
                               const float* __restrict__ asrc,
                               const float* __restrict__ adst,
                               const float* __restrict__ m,
                               const float* __restrict__ ssum,
                               const float* __restrict__ feat,
                               float* __restrict__ agg) {
    constexpr int GPH = C / CG;       // groups per head
    constexpr int TPE = H * GPH;      // threads per edge
    long long tid = blockIdx.x * (long long)blockDim.x + threadIdx.x;
    long long i = tid / TPE;
    if (i >= E) return;
    const int rem = (int)(tid % TPE);
    const int h   = rem / GPH;
    const int c0  = (rem % GPH) * CG;
    const long long s = ei[i], d = ei[E + i];
    float e = asrc[s * H + h] + adst[d * H + h];
    e = e > 0.f ? e : NEG_SLOPE * e;
    const float alpha = __expf(e - m[d * H + h]) / (ssum[d * H + h] + 1e-16f);
    const float* __restrict__ fsrc = feat + ((size_t)s * H + h) * C + c0;
    float* __restrict__ ad = agg + ((size_t)d * H + h) * C + c0;
#pragma unroll
    for (int c = 0; c < CG; ++c)
        atomicAdd(&ad[c], fsrc[c] * alpha);
}

// ---------------------------------------------------------------------------
__global__ void bias_elu(float* x, const float* __restrict__ bias,
                         long long total, int F) {
    long long i = blockIdx.x * (long long)blockDim.x + threadIdx.x;
    if (i >= total) return;
    float v = x[i] + bias[(int)(i % F)];
    x[i] = v > 0.f ? v : (__expf(v) - 1.f);
}

__global__ void bias_add(float* x, const float* __restrict__ bias,
                         long long total, int F) {
    long long i = blockIdx.x * (long long)blockDim.x + threadIdx.x;
    if (i >= total) return;
    x[i] += bias[(int)(i % F)];
}

// ---------------------------------------------------------------------------
static inline unsigned gridFor(long long total, int block) {
    return (unsigned)((total + block - 1) / block);
}

extern "C" void kernel_launch(void* const* d_in, const int* in_sizes, int n_in,
                              void* d_out, int out_size, void* d_ws, size_t ws_size,
                              hipStream_t stream) {
    const float*     x    = (const float*)d_in[0];
    const long long* ei   = (const long long*)d_in[1];
    const float*     W1   = (const float*)d_in[2];
    const float*     as1  = (const float*)d_in[3];
    const float*     ad1  = (const float*)d_in[4];
    const float*     b1   = (const float*)d_in[5];
    const float*     W2   = (const float*)d_in[6];
    const float*     as2  = (const float*)d_in[7];
    const float*     ad2  = (const float*)d_in[8];
    const float*     b2   = (const float*)d_in[9];
    float*           out  = (float*)d_out;

    const int       N = in_sizes[0] / 128;      // 100000
    const long long E = in_sizes[1] / 2;        // 1600000
    constexpr int F1 = 128, H1 = 8, C1 = 16;    // layer 1: 8 heads x 16
    constexpr int C2 = 64;                      // layer 2: 1 head  x 64

    // workspace carve-up (floats)
    float* w = (float*)d_ws;
    float* h1    = w;                 w += (size_t)N * F1;   // GEMM1 out
    float* agg1  = w;                 w += (size_t)N * F1;   // layer1 agg -> ELU in place
    float* h2    = w;                 w += (size_t)N * C2;   // GEMM2 out
    float* asrc1 = w;                 w += (size_t)N * H1;
    float* adst1 = w;                 w += (size_t)N * H1;
    float* m1    = w;                 w += (size_t)N * H1;
    float* s1    = w;                 w += (size_t)N * H1;
    float* asrc2 = w;                 w += (size_t)N;
    float* adst2 = w;                 w += (size_t)N;
    float* m2    = w;                 w += (size_t)N;
    float* s2    = w;                 w += (size_t)N;

    const int BLK = 256;
    const float NEG_INF = -__builtin_huge_valf();

    // ---------------- init accumulators ----------------
    fillf<<<gridFor((long long)N * H1, BLK), BLK, 0, stream>>>(m1, (long long)N * H1, NEG_INF);
    fillf<<<gridFor((long long)N * H1, BLK), BLK, 0, stream>>>(s1, (long long)N * H1, 0.f);
    fillf<<<gridFor((long long)N * F1, BLK), BLK, 0, stream>>>(agg1, (long long)N * F1, 0.f);
    fillf<<<gridFor((long long)N, BLK), BLK, 0, stream>>>(m2, (long long)N, NEG_INF);
    fillf<<<gridFor((long long)N, BLK), BLK, 0, stream>>>(s2, (long long)N, 0.f);
    fillf<<<gridFor((long long)N * C2, BLK), BLK, 0, stream>>>(out, (long long)N * C2, 0.f);

    // ---------------- layer 1 ----------------
    {
        int strips = (N + 15) / 16;
        int blocks = (strips + 7) / 8;          // 8 waves/block, 16 rows/wave
        gemm_wmma<F1><<<blocks, BLK, 0, stream>>>(x, W1, h1, N);
    }
    attn_proj<H1, C1><<<gridFor((long long)N * H1, BLK), BLK, 0, stream>>>(
        h1, as1, ad1, asrc1, adst1, N);
    edge_max<H1><<<gridFor(E, BLK), BLK, 0, stream>>>(ei, E, asrc1, adst1, m1);
    edge_sum<H1><<<gridFor(E, BLK), BLK, 0, stream>>>(ei, E, asrc1, adst1, m1, s1);
    edge_aggregate<H1, C1, 4><<<gridFor(E * (H1 * C1 / 4), BLK), BLK, 0, stream>>>(
        ei, E, asrc1, adst1, m1, s1, h1, agg1);
    bias_elu<<<gridFor((long long)N * F1, BLK), BLK, 0, stream>>>(
        agg1, b1, (long long)N * F1, F1);

    // ---------------- layer 2 ----------------
    {
        int strips = (N + 15) / 16;
        int blocks = (strips + 7) / 8;
        gemm_wmma<C2><<<blocks, BLK, 0, stream>>>(agg1, W2, h2, N);
    }
    attn_proj<1, C2><<<gridFor((long long)N, BLK), BLK, 0, stream>>>(
        h2, as2, ad2, asrc2, adst2, N);
    edge_max<1><<<gridFor(E, BLK), BLK, 0, stream>>>(ei, E, asrc2, adst2, m2);
    edge_sum<1><<<gridFor(E, BLK), BLK, 0, stream>>>(ei, E, asrc2, adst2, m2, s2);
    edge_aggregate<1, C2, 4><<<gridFor(E * (C2 / 4), BLK), BLK, 0, stream>>>(
        ei, E, asrc2, adst2, m2, s2, h2, out);
    bias_add<<<gridFor((long long)N * C2, BLK), BLK, 0, stream>>>(
        out, b2, (long long)N * C2, C2);
}
